// GIN_75204877353218
// MI455X (gfx1250) — compile-verified, matrix operand order checked by reading, not run on hardware
//
#include <hip/hip_runtime.h>
#include <stdint.h>

// ---------------------------------------------------------------------------
// GIN forward for MI455X (gfx1250, wave32).
//   agg = scatter_add(x[src] -> dst)        : global_atomic_add_f32 (L2-resident),
//                                             edge indices scalarized -> s_load_b64
//   h   = relu(lin2(relu(bn(lin1(x+agg))))) : v_wmma_f32_16x16x32_bf16, fp32 accum
//   (x2 layers) then head: relu(lin)+lin    : same WMMA path
// BN is folded into per-feature scale/bias; weights pre-converted to bf16.
// ---------------------------------------------------------------------------

typedef __bf16 bf16_t;
typedef __attribute__((ext_vector_type(16))) __bf16 v16bf;
typedef __attribute__((ext_vector_type(8)))  __bf16 v8bf;
typedef __attribute__((ext_vector_type(8)))  float  v8f;
typedef __attribute__((ext_vector_type(4)))  float  v4f;

#define D_FEAT 128

// ---- packing helpers ------------------------------------------------------

__device__ __forceinline__ v16bf pack16(v8bf lo, v8bf hi) {
    v16bf r;
#pragma unroll
    for (int i = 0; i < 8; ++i) { r[i] = lo[i]; r[i + 8] = hi[i]; }
    return r;
}

__device__ __forceinline__ v16bf cvt16(v4f a0, v4f a1, v4f a2, v4f a3) {
    v16bf r;
#pragma unroll
    for (int i = 0; i < 4; ++i) {
        r[i]      = (bf16_t)a0[i];
        r[4 + i]  = (bf16_t)a1[i];
        r[8 + i]  = (bf16_t)a2[i];
        r[12 + i] = (bf16_t)a3[i];
    }
    return r;
}

// A-matrix 16x32 bf16 chunk, ISA layout:
//   lanes 0-15: row=lane, K = kc*32 + {0..7} (v0-3) and {16..23} (v4-7)
//   lanes16-31: row=lane-16, K = kc*32 + {8..15} and {24..31}
__device__ __forceinline__ v16bf load_a_sum(const float* x, const float* agg,
                                            long long row, int kc, int lane) {
    const int half = lane >> 4;
    const long long off = row * D_FEAT + kc * 32 + half * 8;
    const v4f* px = (const v4f*)(x + off);
    const v4f* pa = (const v4f*)(agg + off);
    v4f a0 = px[0] + pa[0];
    v4f a1 = px[1] + pa[1];
    v4f a2 = px[4] + pa[4];
    v4f a3 = px[5] + pa[5];
    return cvt16(a0, a1, a2, a3);
}

__device__ __forceinline__ v16bf load_a_plain(const float* x, long long row,
                                              int kc, int lane) {
    const int half = lane >> 4;
    const long long off = row * D_FEAT + kc * 32 + half * 8;
    const v4f* px = (const v4f*)(x + off);
    return cvt16(px[0], px[1], px[4], px[5]);
}

// B-matrix 32x16 bf16 chunk for h = x @ W^T: lane n holds column n of B,
// i.e. row (nt*16 + lane&15) of W  -> contiguous bf16 loads.
__device__ __forceinline__ v16bf load_b_w(const bf16_t* W, int nt, int kc, int lane) {
    const int n = nt * 16 + (lane & 15);
    const int half = lane >> 4;
    const bf16_t* p = W + n * D_FEAT + kc * 32 + half * 8;
    return pack16(*(const v8bf*)p, *(const v8bf*)(p + 16));
}

__device__ __forceinline__ v16bf load_a_lds(const bf16_t* L, int kc, int lane) {
    const int m = lane & 15;
    const int half = lane >> 4;
    const bf16_t* p = L + m * D_FEAT + kc * 32 + half * 8;
    return pack16(*(const v8bf*)p, *(const v8bf*)(p + 16));
}

// ---- prep kernels ---------------------------------------------------------

__global__ void cvt_bf16_kernel(const float* __restrict__ s, bf16_t* __restrict__ d, int n) {
    int i = blockIdx.x * blockDim.x + threadIdx.x;
    if (i < n) d[i] = (bf16_t)s[i];
}

// fold BN(eval) into scale/bias applied after the raw GEMM (bias absorbed):
//   y = (xW^T + ba - rm) * g * rsqrt(rv+eps) + be  =  (xW^T)*s + (s*(ba-rm)+be)
__global__ void fold_bn_kernel(const float* __restrict__ g,  const float* __restrict__ be,
                               const float* __restrict__ rm, const float* __restrict__ rv,
                               const float* __restrict__ ba,
                               float* __restrict__ scl, float* __restrict__ bias) {
    int i = threadIdx.x;
    float s = g[i] * rsqrtf(rv[i] + 1e-5f);
    scl[i]  = s;
    bias[i] = s * (ba[i] - rm[i]) + be[i];
}

// vectorized zero: n must be a multiple of 4 (N*128 is)
__global__ void zero_f32v4_kernel(v4f* __restrict__ p, long long n4) {
    long long i = (long long)blockIdx.x * blockDim.x + threadIdx.x;
    if (i < n4) p[i] = (v4f){0.0f, 0.0f, 0.0f, 0.0f};
}

// ---- edge aggregation: agg[dst] += feat[src] ------------------------------
// One wave per edge (32 lanes x float4 = 128 channels). The edge id is
// wave-uniform; readfirstlane makes that provable so the two int64 index
// loads become scalar s_load_b64 instead of 32 redundant vector loads.
__global__ __launch_bounds__(256)
void scatter_add_kernel(const float* __restrict__ feat, const long long* __restrict__ ei,
                        float* __restrict__ agg, long long E) {
    long long t = (long long)blockIdx.x * blockDim.x + threadIdx.x;
    int e = __builtin_amdgcn_readfirstlane((int)(t >> 5));   // wave-uniform edge id
    int c = (int)(t & 31) * 4;
    if (e >= (int)E) return;
    long long s = ei[e];          // scalar (SMEM) load
    long long d = ei[E + e];      // scalar (SMEM) load
    v4f v = *(const v4f*)(feat + s * D_FEAT + c);
    float* p = agg + d * D_FEAT + c;
    atomicAdd(p + 0, v[0]);
    atomicAdd(p + 1, v[1]);
    atomicAdd(p + 2, v[2]);
    atomicAdd(p + 3, v[3]);
}

// ---- fused GIN MLP: out = relu( relu(bn(lin_a(x+agg))) @ Wb^T + bb ) ------
// 4 waves/block, 1 wave = 16-node tile. x may alias out (per-row in-place).
__global__ __launch_bounds__(128)
void gin_mlp_wmma(const float* x, const float* __restrict__ agg,
                  const bf16_t* __restrict__ Wa, const float* __restrict__ sclA,
                  const float* __restrict__ bsA,
                  const bf16_t* __restrict__ Wb, const float* __restrict__ bsB,
                  float* out, int n_nodes) {
    __shared__ __align__(16) bf16_t lds_h[4][16 * D_FEAT];
    const int wid  = threadIdx.x >> 5;
    const int lane = threadIdx.x & 31;
    const int ntiles = (n_nodes + 15) >> 4;
    const int tile = blockIdx.x * 4 + wid;
    if (tile >= ntiles) return;
    const long long row0 = (long long)tile * 16;
    const int m = lane & 15, half = lane >> 4;
    long long arow = row0 + m;
    if (arow >= n_nodes) arow = n_nodes - 1;

    v16bf a[4];
#pragma unroll
    for (int kc = 0; kc < 4; ++kc) a[kc] = load_a_sum(x, agg, arow, kc, lane);

    // GEMM1: 8 output tiles of 16x16, K=128 in 4 bf16 chunks
    v8f c[8];
#pragma unroll
    for (int nt = 0; nt < 8; ++nt) {
        v8f acc = {};
#pragma unroll
        for (int kc = 0; kc < 4; ++kc)
            acc = __builtin_amdgcn_wmma_f32_16x16x32_bf16(
                false, a[kc], false, load_b_w(Wa, nt, kc, lane),
                (short)0, acc, false, false);
        c[nt] = acc;
    }

    // epilogue1: folded-BN scale/bias + relu, reshape D-layout -> LDS rows
    bf16_t* L = lds_h[wid];
#pragma unroll
    for (int nt = 0; nt < 8; ++nt) {
        int n = nt * 16 + m;
        float s = sclA[n], b = bsA[n];
#pragma unroll
        for (int r = 0; r < 8; ++r) {
            float v = c[nt][r] * s + b;
            v = v > 0.0f ? v : 0.0f;
            L[(r + half * 8) * D_FEAT + n] = (bf16_t)v;
        }
    }
    asm volatile("s_wait_dscnt 0x0" ::: "memory");

    // GEMM2 from LDS-staged activations
#pragma unroll
    for (int kc = 0; kc < 4; ++kc) a[kc] = load_a_lds(L, kc, lane);

#pragma unroll
    for (int nt = 0; nt < 8; ++nt) {
        v8f acc = {};
#pragma unroll
        for (int kc = 0; kc < 4; ++kc)
            acc = __builtin_amdgcn_wmma_f32_16x16x32_bf16(
                false, a[kc], false, load_b_w(Wb, nt, kc, lane),
                (short)0, acc, false, false);
        int n = nt * 16 + m;
        float b = bsB[n];
#pragma unroll
        for (int r = 0; r < 8; ++r) {
            long long rr = row0 + r + half * 8;
            if (rr < n_nodes) {
                float v = acc[r] + b;
                out[rr * D_FEAT + n] = v > 0.0f ? v : 0.0f;
            }
        }
    }
}

// ---- head: out = relu(h @ Wl1^T + b1) @ Wl2^T + b2   (128 -> 128 -> 64) ---
__global__ __launch_bounds__(128)
void head_wmma(const float* __restrict__ h,
               const bf16_t* __restrict__ W1, const float* __restrict__ b1,
               const bf16_t* __restrict__ W2, const float* __restrict__ b2,
               float* __restrict__ out, int n_nodes) {
    __shared__ __align__(16) bf16_t lds_h[4][16 * D_FEAT];
    const int wid  = threadIdx.x >> 5;
    const int lane = threadIdx.x & 31;
    const int ntiles = (n_nodes + 15) >> 4;
    const int tile = blockIdx.x * 4 + wid;
    if (tile >= ntiles) return;
    const long long row0 = (long long)tile * 16;
    const int m = lane & 15, half = lane >> 4;
    long long arow = row0 + m;
    if (arow >= n_nodes) arow = n_nodes - 1;

    v16bf a[4];
#pragma unroll
    for (int kc = 0; kc < 4; ++kc) a[kc] = load_a_plain(h, arow, kc, lane);

    bf16_t* L = lds_h[wid];
#pragma unroll
    for (int nt = 0; nt < 8; ++nt) {
        v8f acc = {};
#pragma unroll
        for (int kc = 0; kc < 4; ++kc)
            acc = __builtin_amdgcn_wmma_f32_16x16x32_bf16(
                false, a[kc], false, load_b_w(W1, nt, kc, lane),
                (short)0, acc, false, false);
        int n = nt * 16 + m;
        float b = b1[n];
#pragma unroll
        for (int r = 0; r < 8; ++r) {
            float v = acc[r] + b;
            v = v > 0.0f ? v : 0.0f;
            L[(r + half * 8) * D_FEAT + n] = (bf16_t)v;
        }
    }
    asm volatile("s_wait_dscnt 0x0" ::: "memory");

#pragma unroll
    for (int kc = 0; kc < 4; ++kc) a[kc] = load_a_lds(L, kc, lane);

#pragma unroll
    for (int nt = 0; nt < 4; ++nt) {  // Cout = 64
        v8f acc = {};
#pragma unroll
        for (int kc = 0; kc < 4; ++kc)
            acc = __builtin_amdgcn_wmma_f32_16x16x32_bf16(
                false, a[kc], false, load_b_w(W2, nt, kc, lane),
                (short)0, acc, false, false);
        int n = nt * 16 + m;
        float b = b2[n];
#pragma unroll
        for (int r = 0; r < 8; ++r) {
            long long rr = row0 + r + half * 8;
            if (rr < n_nodes) out[rr * 64 + n] = acc[r] + b;
        }
    }
}

// ---------------------------------------------------------------------------

extern "C" void kernel_launch(void* const* d_in, const int* in_sizes, int n_in,
                              void* d_out, int out_size, void* d_ws, size_t ws_size,
                              hipStream_t stream) {
    const float*     x   = (const float*)d_in[0];
    const long long* ei  = (const long long*)d_in[1];   // int64 edge_index [2,E]
    const float* w1a = (const float*)d_in[2],  *b1a = (const float*)d_in[3];
    const float* g1  = (const float*)d_in[4],  *be1 = (const float*)d_in[5];
    const float* rm1 = (const float*)d_in[6],  *rv1 = (const float*)d_in[7];
    const float* w1b = (const float*)d_in[8],  *b1b = (const float*)d_in[9];
    const float* w2a = (const float*)d_in[10], *b2a = (const float*)d_in[11];
    const float* g2  = (const float*)d_in[12], *be2 = (const float*)d_in[13];
    const float* rm2 = (const float*)d_in[14], *rv2 = (const float*)d_in[15];
    const float* w2b = (const float*)d_in[16], *b2b = (const float*)d_in[17];
    const float* wl1 = (const float*)d_in[18], *bl1 = (const float*)d_in[19];
    const float* wl2 = (const float*)d_in[20], *bl2 = (const float*)d_in[21];

    const long long Nn = in_sizes[0] / D_FEAT;   // 50000 nodes
    const long long E  = in_sizes[1] / 2;        // 800000 edges
    (void)n_in; (void)out_size; (void)ws_size;

    // workspace carve-up (256B aligned)
    char* wsp = (char*)d_ws;
    size_t off = 0;
    auto alloc = [&](size_t bytes) -> char* {
        char* p = wsp + off;
        off += (bytes + 255) & ~(size_t)255;
        return p;
    };
    float*  agg  = (float*)alloc((size_t)Nn * D_FEAT * 4);
    float*  hbuf = (float*)alloc((size_t)Nn * D_FEAT * 4);
    bf16_t* W1a  = (bf16_t*)alloc(128 * 128 * 2);
    bf16_t* W1b  = (bf16_t*)alloc(128 * 128 * 2);
    bf16_t* W2a  = (bf16_t*)alloc(128 * 128 * 2);
    bf16_t* W2b  = (bf16_t*)alloc(128 * 128 * 2);
    bf16_t* WL1  = (bf16_t*)alloc(128 * 128 * 2);
    bf16_t* WL2  = (bf16_t*)alloc(64 * 128 * 2);
    float*  s1   = (float*)alloc(128 * 4);
    float*  bb1  = (float*)alloc(128 * 4);
    float*  s2   = (float*)alloc(128 * 4);
    float*  bb2  = (float*)alloc(128 * 4);

    // --- prep: bf16 weights + folded BN ---
    cvt_bf16_kernel<<<64, 256, 0, stream>>>(w1a, W1a, 128 * 128);
    cvt_bf16_kernel<<<64, 256, 0, stream>>>(w1b, W1b, 128 * 128);
    cvt_bf16_kernel<<<64, 256, 0, stream>>>(w2a, W2a, 128 * 128);
    cvt_bf16_kernel<<<64, 256, 0, stream>>>(w2b, W2b, 128 * 128);
    cvt_bf16_kernel<<<64, 256, 0, stream>>>(wl1, WL1, 128 * 128);
    cvt_bf16_kernel<<<32, 256, 0, stream>>>(wl2, WL2, 64 * 128);
    fold_bn_kernel<<<1, 128, 0, stream>>>(g1, be1, rm1, rv1, b1a, s1, bb1);
    fold_bn_kernel<<<1, 128, 0, stream>>>(g2, be2, rm2, rv2, b2a, s2, bb2);

    const long long nf4 = Nn * D_FEAT / 4;
    const int zeroBlocks = (int)((nf4 + 255) / 256);
    const int scatBlocks = (int)((E * 32 + 255) / 256);
    const int ntiles     = (int)((Nn + 15) / 16);
    const int mlpBlocks  = (ntiles + 3) / 4;

    // --- layer 1 ---
    zero_f32v4_kernel<<<zeroBlocks, 256, 0, stream>>>((v4f*)agg, nf4);
    scatter_add_kernel<<<scatBlocks, 256, 0, stream>>>(x, ei, agg, E);
    gin_mlp_wmma<<<mlpBlocks, 128, 0, stream>>>(x, agg, W1a, s1, bb1, W1b, b1b,
                                                hbuf, (int)Nn);
    // --- layer 2 (in-place over hbuf) ---
    zero_f32v4_kernel<<<zeroBlocks, 256, 0, stream>>>((v4f*)agg, nf4);
    scatter_add_kernel<<<scatBlocks, 256, 0, stream>>>(hbuf, ei, agg, E);
    gin_mlp_wmma<<<mlpBlocks, 128, 0, stream>>>(hbuf, agg, W2a, s2, bb2, W2b, b2b,
                                                hbuf, (int)Nn);
    // --- head ---
    head_wmma<<<mlpBlocks, 128, 0, stream>>>(hbuf, WL1, bl1, WL2, bl2,
                                             (float*)d_out, (int)Nn);
}
